// HierarchicalGNNBlock_22368189678471
// MI455X (gfx1250) — compile-verified, matrix operand order checked by reading, not run on hardware
//
#include <hip/hip_runtime.h>
#include <hip/hip_bf16.h>

// ---------------- problem constants (from reference) ----------------
#define NNODES     100000
#define NEDGE      600000
#define NCLUST     1024
#define LATENT     128
#define HIDDEN     256
#define EMB        24
#define KSUP       8
#define KBIP       4
#define NSEDGE     (NCLUST * KSUP * 2)   // 16384 (symmetric)
#define NBEDGE     (NNODES * KBIP)       // 400000

typedef _Float16 v16h __attribute__((ext_vector_type(16)));
typedef _Float16 v8h  __attribute__((ext_vector_type(8)));
typedef float    v8f  __attribute__((ext_vector_type(8)));

// =====================================================================
//  WMMA MLP layer:  Y = GELU(LayerNorm(X @ W + b) * g + be) [+ residual]
//  block = (32, NT); one wave per 16x16 output tile; block owns a
//  16-row stripe and all N columns (LayerNorm needs the full row).
//  A comes from LDS (f16, staged once per block) or dense f16 global;
//  B comes from pre-transposed f16 weights Wt[Npad][K] (zero-padded).
// =====================================================================

struct Cat3 {   // up to 3 concatenated sources, each 128 features wide
    const float* p0; const int* i0;
    const float* p1; const int* i1;
    const float* p2; const int* i2;
};

// shared tail: K-loop + bias + LayerNorm + GELU + store
__device__ __forceinline__ void mlp_body(
    const _Float16* xrow,        // this lane's A row base (K contiguous f16)
    const _Float16* wrow,        // this lane's B column base (K contiguous f16)
    float* sD, float* smu, float* srs,
    int M, int K,
    const float* bias, const float* gam, const float* bet,
    int Nout, int NT,
    _Float16* Yh, float* Yf, int ostride, const float* res, int rstride)
{
    const int lane = threadIdx.x;
    const int wv   = threadIdx.y;
    const int m0   = blockIdx.x * 16;
    const int ml   = lane & 15;
    const int kh   = lane >> 4;
    const int ldn  = NT * 16 + 1;
    const int nb   = wv * 16 + ml;       // output column owned by this lane

    v8f acc = {};
    for (int k0 = 0; k0 < K; k0 += 32) {
        v8h alo = *(const v8h*)(xrow + k0 + kh * 8);
        v8h ahi = *(const v8h*)(xrow + k0 + 16 + kh * 8);
        v8h blo = *(const v8h*)(wrow + k0 + kh * 8);
        v8h bhi = *(const v8h*)(wrow + k0 + 16 + kh * 8);
        v16h a = __builtin_shufflevector(alo, ahi, 0,1,2,3,4,5,6,7,8,9,10,11,12,13,14,15);
        v16h b = __builtin_shufflevector(blo, bhi, 0,1,2,3,4,5,6,7,8,9,10,11,12,13,14,15);
        acc = __builtin_amdgcn_wmma_f32_16x16x32_f16(
            false, a, false, b, (short)0, acc, false, false);
    }

    // D tile -> LDS (+bias).  D layout: VGPR r -> row (kh*8+r), col nb.
    if (nb < Nout) {
        float bb = bias[nb];
#pragma unroll
        for (int r = 0; r < 8; ++r) sD[(kh * 8 + r) * ldn + nb] = acc[r] + bb;
    }
    __syncthreads();

    const int tid = wv * 32 + lane, nth = NT * 32;
    for (int row = tid; row < 16; row += nth) {
        float s = 0.f, s2 = 0.f;
        for (int j = 0; j < Nout; ++j) { float v = sD[row * ldn + j]; s += v; s2 += v * v; }
        float mu = s / Nout;
        float var = s2 / Nout - mu * mu;
        smu[row] = mu;
        srs[row] = rsqrtf(var + 1e-5f);
    }
    __syncthreads();

    const int total = 16 * Nout;
    for (int t = tid; t < total; t += nth) {
        int row = t / Nout, j = t - row * Nout;
        int mm = m0 + row;
        if (mm >= M) continue;
        float v = (sD[row * ldn + j] - smu[row]) * srs[row] * gam[j] + bet[j];
        // GELU (tanh approximation, jax default)
        float u = 0.7978845608028654f * (v + 0.044715f * v * v * v);
        float gel = 0.5f * v * (1.0f + tanhf(u));
        if (Yh) {
            Yh[(size_t)mm * ostride + j] = (_Float16)gel;
        } else {
            float o = gel + (res ? res[(size_t)mm * rstride + j] : 0.0f);
            Yf[(size_t)mm * ostride + j] = o;
        }
    }
}

// first layer: gather-concat f32 sources -> LDS f16 tile, then WMMA.
// always launched with blockDim=(32,16)=512 threads, Nout=HIDDEN.
__global__ void __launch_bounds__(512) k_mlp_cat(
    Cat3 c, int M, int K, const _Float16* Wt,
    const float* bias, const float* gam, const float* bet,
    _Float16* Yh)
{
    extern __shared__ char smraw[];
    _Float16* Xs = (_Float16*)smraw;                    // [16][K] f16
    float* sD  = (float*)(smraw + (size_t)16 * K * 2);  // 16*K*2 is 16B-multiple
    float* smu = sD + 16 * (16 * 16 + 1);
    float* srs = smu + 16;

    const int tid  = threadIdx.y * 32 + threadIdx.x;    // 0..511
    const int m0   = blockIdx.x * 16;
    const int nseg = K >> 7;                            // segments of 128

    // stage: one float4 chunk per thread per segment (16 rows x 32 chunks)
    const int row = tid >> 5;
    const int c4  = (tid & 31) << 2;
    int m = m0 + row; if (m >= M) m = M - 1;
    for (int seg = 0; seg < nseg; ++seg) {
        const float* p  = seg == 0 ? c.p0 : (seg == 1 ? c.p1 : c.p2);
        const int*   ix = seg == 0 ? c.i0 : (seg == 1 ? c.i1 : c.i2);
        int r = ix ? ix[m] : m;
        float4 v = *(const float4*)(p + (size_t)r * 128 + c4);
        _Float16* dst = Xs + row * K + (seg << 7) + c4;
        dst[0] = (_Float16)v.x; dst[1] = (_Float16)v.y;
        dst[2] = (_Float16)v.z; dst[3] = (_Float16)v.w;
    }
    __syncthreads();

    const _Float16* xrow = Xs + (threadIdx.x & 15) * K;
    const _Float16* wrow = Wt + (size_t)(threadIdx.y * 16 + (threadIdx.x & 15)) * K;
    mlp_body(xrow, wrow, sD, smu, srs, M, K, bias, gam, bet,
             HIDDEN, 16, Yh, nullptr, HIDDEN, nullptr, 0);
}

// mid/last layer: dense f16 input from global
__global__ void __launch_bounds__(512) k_mlp_h16(
    const _Float16* X, int ldx, int M, int K, const _Float16* Wt,
    const float* bias, const float* gam, const float* bet,
    int Nout, int NT, _Float16* Yh, float* Yf, int ostride,
    const float* res, int rstride)
{
    extern __shared__ char smraw[];
    float* sD  = (float*)smraw;
    float* smu = sD + 16 * (NT * 16 + 1);
    float* srs = smu + 16;

    const int m0 = blockIdx.x * 16;
    int m = m0 + (threadIdx.x & 15); if (m >= M) m = M - 1;
    const _Float16* xrow = X + (size_t)m * ldx;
    const _Float16* wrow = Wt + (size_t)(threadIdx.y * 16 + (threadIdx.x & 15)) * K;
    mlp_body(xrow, wrow, sD, smu, srs, M, K, bias, gam, bet,
             Nout, NT, Yh, Yf, ostride, res, rstride);
}

// weight prep: W[K][N] f32 row-major -> Wt[Npad][K] f16 (zero pad n>=N)
__global__ void k_prep_wt(const float* W, int K, int N, int Npad, _Float16* Wt)
{
    int t = blockIdx.x * blockDim.x + threadIdx.x;
    if (t >= Npad * K) return;
    int n = t / K, k = t - n * K;
    Wt[t] = (_Float16)(n < N ? W[(size_t)k * N + n] : 0.0f);
}

// =====================================================================
//  Graph construction / aggregation kernels
// =====================================================================

__global__ void k_cluster_accum(const float* emb, const int* cl,
                                float* sums, float* cnt, int n)
{
    int i = blockIdx.x * blockDim.x + threadIdx.x;
    if (i >= n) return;
    int c = cl[i];
    if (c < 0 || c >= NCLUST) return;
#pragma unroll
    for (int j = 0; j < EMB; ++j)
        atomicAdd(&sums[c * EMB + j], emb[(size_t)i * EMB + j]);
    atomicAdd(&cnt[c], 1.0f);
}

__global__ void k_cluster_final(float* means, const float* cnt)
{
    int c = blockIdx.x * blockDim.x + threadIdx.x;
    if (c >= NCLUST) return;
    float inv = 1.0f / fmaxf(cnt[c], 1.0f);
    float v[EMB]; float nrm = 0.f;
#pragma unroll
    for (int j = 0; j < EMB; ++j) { v[j] = means[c * EMB + j] * inv; nrm += v[j] * v[j]; }
    float r = 1.0f / fmaxf(sqrtf(nrm), 1e-12f);
#pragma unroll
    for (int j = 0; j < EMB; ++j) means[c * EMB + j] = v[j] * r;
}

template <int TK>
__device__ __forceinline__ void topk_ins(float (&bv)[TK], int (&bi)[TK], float s, int c)
{
    if (!(s > bv[TK - 1])) return;     // strict > : keeps earliest index on ties
    bv[TK - 1] = s; bi[TK - 1] = c;
#pragma unroll
    for (int q = TK - 1; q > 0; --q)
        if (bv[q] > bv[q - 1]) {
            float tv = bv[q]; bv[q] = bv[q - 1]; bv[q - 1] = tv;
            int   ti = bi[q]; bi[q] = bi[q - 1]; bi[q - 1] = ti;
        }
}

__global__ void __launch_bounds__(256) k_bip_knn(
    const float* emb, const float* means, int* bidx, float* blik)
{
    __shared__ float sm[256 * EMB];
    int i = blockIdx.x * 256 + threadIdx.x;
    bool act = i < NNODES;
    float er[EMB];
#pragma unroll
    for (int j = 0; j < EMB; ++j) er[j] = 0.f;
    if (act)
#pragma unroll
        for (int j = 0; j < EMB; ++j) er[j] = emb[(size_t)i * EMB + j];
    float bv[KBIP]; int bi[KBIP];
#pragma unroll
    for (int q = 0; q < KBIP; ++q) { bv[q] = -3.402823e38f; bi[q] = 0; }
    for (int c0 = 0; c0 < NCLUST; c0 += 256) {
        __syncthreads();
        for (int t = threadIdx.x; t < 256 * EMB; t += 256)
            sm[t] = means[(size_t)c0 * EMB + t];
        __syncthreads();
        if (act)
            for (int c = 0; c < 256; ++c) {
                float s = 0.f;
#pragma unroll
                for (int j = 0; j < EMB; ++j) s += er[j] * sm[c * EMB + j];
                topk_ins<KBIP>(bv, bi, s, c0 + c);
            }
    }
    if (act)
#pragma unroll
        for (int q = 0; q < KBIP; ++q) {
            bidx[i * KBIP + q] = bi[q];
            blik[i * KBIP + q] = bv[q];
        }
}

__global__ void __launch_bounds__(256) k_super_knn(
    const float* means, int* sg0, int* sg1, float* lik)
{
    __shared__ float sm[256 * EMB];
    int r = blockIdx.x * 256 + threadIdx.x;    // grid=4 -> r in [0,1024)
    float er[EMB];
#pragma unroll
    for (int j = 0; j < EMB; ++j) er[j] = means[(size_t)r * EMB + j];
    float bv[KSUP]; int bi[KSUP];
#pragma unroll
    for (int q = 0; q < KSUP; ++q) { bv[q] = -3.402823e38f; bi[q] = 0; }
    for (int c0 = 0; c0 < NCLUST; c0 += 256) {
        __syncthreads();
        for (int t = threadIdx.x; t < 256 * EMB; t += 256)
            sm[t] = means[(size_t)c0 * EMB + t];
        __syncthreads();
        for (int c = 0; c < 256; ++c) {
            float s = 0.f;
#pragma unroll
            for (int j = 0; j < EMB; ++j) s += er[j] * sm[c * EMB + j];
            topk_ins<KSUP>(bv, bi, s, c0 + c);
        }
    }
    const int H = NCLUST * KSUP;       // 8192
#pragma unroll
    for (int q = 0; q < KSUP; ++q) {
        int e = r * KSUP + q;
        sg0[e] = r;         sg1[e] = bi[q];   lik[e] = bv[q];
        sg0[H + e] = bi[q]; sg1[H + e] = r;   lik[H + e] = bv[q]; // symmetric
    }
}

__global__ void __launch_bounds__(256) k_reduce2(const float* x, int n, float* out2)
{
    __shared__ float s1[256], s2[256];
    int tid = threadIdx.x;
    float a = 0.f, b = 0.f;
    for (int i = blockIdx.x * 256 + tid; i < n; i += gridDim.x * 256) {
        float v = x[i]; a += v; b += v * v;
    }
    s1[tid] = a; s2[tid] = b; __syncthreads();
    for (int s = 128; s > 0; s >>= 1) {
        if (tid < s) { s1[tid] += s1[tid + s]; s2[tid] += s2[tid + s]; }
        __syncthreads();
    }
    if (tid == 0) { atomicAdd(&out2[0], s1[0]); atomicAdd(&out2[1], s2[0]); }
}

__global__ void __launch_bounds__(256) k_logits_w(
    const float* lik, int n, const float* red2, int use_sigmoid,
    float* logits, float* wout, float* wsum)
{
    __shared__ float swr[256];
    float mu  = red2[0] / n;
    float var = red2[1] / n - mu * mu;
    float rs  = rsqrtf(var + 1e-5f);
    float acc = 0.f;
    for (int i = blockIdx.x * 256 + threadIdx.x; i < n; i += gridDim.x * 256) {
        float l = (lik[i] - mu) * rs;
        float wv = use_sigmoid ? (1.0f / (1.0f + expf(-l))) : expf(l);
        if (logits) logits[i] = l;
        wout[i] = wv;
        acc += wv;
    }
    swr[threadIdx.x] = acc; __syncthreads();
    for (int s = 128; s > 0; s >>= 1) {
        if (threadIdx.x < s) swr[threadIdx.x] += swr[threadIdx.x + s];
        __syncthreads();
    }
    if (threadIdx.x == 0) atomicAdd(wsum, swr[0]);
}

__global__ void k_w_norm(float* wv, int n, const float* wsum)
{
    int i = blockIdx.x * blockDim.x + threadIdx.x;
    if (i < n) wv[i] *= (float)n / wsum[0];
}

// sn_in = segment_sum(L1norm(nodes)[e>>2] * bip_w[e] -> cluster bidx[e])
__global__ void k_sn_in(const float* nodes0, const int* bidx, const float* bw,
                        float* snin, int ne)
{
    int e = blockIdx.x * blockDim.x + threadIdx.x;
    if (e >= ne) return;
    int src = e >> 2;
    int c   = bidx[e];
    float wv = bw[e];
    const float* xr = nodes0 + (size_t)src * LATENT;
    float l1 = 0.f;
    for (int j = 0; j < LATENT; ++j) l1 += fabsf(xr[j]);
    float sc = wv / fmaxf(l1, 1e-12f);
    for (int j = 0; j < LATENT; ++j)
        atomicAdd(&snin[(size_t)c * LATENT + j], xr[j] * sc);
}

// generic weighted segment-sum scatter over LATENT-dim rows
__global__ void k_scatter_add(const float* src, const int* sidx, int sshift,
                              const float* wv, const int* didx, int dshift,
                              float* outp, long long nfe)
{
    long long gid = (long long)blockIdx.x * blockDim.x + threadIdx.x;
    if (gid >= nfe) return;
    int e = (int)(gid >> 7);
    int f = (int)(gid & 127);
    int sr = sidx ? sidx[e] : (e >> sshift);
    int dr = didx ? didx[e] : (e >> dshift);
    float v = src[(size_t)sr * LATENT + f];
    if (wv) v *= wv[e];
    atomicAdd(&outp[(size_t)dr * LATENT + f], v);
}

__global__ void k_copy_means(const float* means, float* sn)
{
    int t = blockIdx.x * blockDim.x + threadIdx.x;
    if (t >= NCLUST * EMB) return;
    int r = t / EMB, j = t - r * EMB;
    sn[(size_t)r * LATENT + j] = means[t];
}

__global__ void k_bip_out(const int* bidx, float* outp, int ne)
{
    int e = blockIdx.x * blockDim.x + threadIdx.x;
    if (e >= ne) return;
    outp[e]      = (float)(e >> 2);   // bip_graph[0]
    outp[ne + e] = (float)bidx[e];    // bip_graph[1]
}

// =====================================================================
//  Host orchestration
// =====================================================================

static inline unsigned gb(long long n) { return (unsigned)((n + 255) / 256); }

static void run_mlp3(hipStream_t s, int M, const Cat3& c, int K,
                     _Float16* const* Wt, const float* const* Wp,
                     _Float16* H1, _Float16* H2,
                     int NoutF, float* outF, int ostride,
                     const float* res, int rstride)
{
    int mb = M / 16;                   // all M here are multiples of 16
    dim3 blk(32, 16);
    size_t shm1 = (size_t)16 * K * 2 + (size_t)(16 * (16 * 16 + 1) + 32) * sizeof(float);
    k_mlp_cat<<<mb, blk, shm1, s>>>(c, M, K, Wt[0], Wp[1], Wp[2], Wp[3], H1);
    size_t shm2 = (size_t)(16 * (16 * 16 + 1) + 32) * sizeof(float);
    k_mlp_h16<<<mb, blk, shm2, s>>>(H1, HIDDEN, M, HIDDEN, Wt[1], Wp[5], Wp[6], Wp[7],
                                    HIDDEN, 16, H2, nullptr, HIDDEN, nullptr, 0);
    int NT = (NoutF + 15) / 16;
    dim3 blk3(32, NT);
    size_t shm3 = (size_t)(16 * (NT * 16 + 1) + 32) * sizeof(float);
    k_mlp_h16<<<mb, blk3, shm3, s>>>(H2, HIDDEN, M, HIDDEN, Wt[2], Wp[9], Wp[10], Wp[11],
                                     NoutF, NT, nullptr, outF, ostride, res, rstride);
}

extern "C" void kernel_launch(void* const* d_in, const int* in_sizes, int n_in,
                              void* d_out, int out_size, void* d_ws, size_t ws_size,
                              hipStream_t stream)
{
    (void)in_sizes; (void)n_in; (void)out_size; (void)ws_size;

    const float* emb      = (const float*)d_in[1];
    const float* nodes_in = (const float*)d_in[2];
    const float* edges_in = (const float*)d_in[3];
    const int*   graph    = (const int*)d_in[4];
    const int*   gr0 = graph;
    const int*   gr1 = graph + NEDGE;
    const int*   clusters = (const int*)d_in[5];

    // raw param access: d_in[6 + mlp*12 + layer*4 + {W,b,g,be}]
    auto getW = [&](int mlp, const float** arr) {
        for (int l = 0; l < 3; ++l)
            for (int it = 0; it < 4; ++it)
                arr[l * 4 + it] = (const float*)d_in[6 + mlp * 12 + l * 4 + it];
    };

    // ---- workspace carve ----
    char* base = (char*)d_ws; size_t off = 0;
    auto alloc = [&](size_t bytes) -> void* {
        void* p = base + off;
        off = (off + bytes + 255) & ~(size_t)255;
        return p;
    };
    float* means   = (float*)alloc((size_t)NCLUST * EMB * 4);
    float* counts  = (float*)alloc((size_t)NCLUST * 4);
    float* red     = (float*)alloc(64);
    int*   sg0     = (int*)  alloc((size_t)NSEDGE * 4);
    int*   sg1     = (int*)  alloc((size_t)NSEDGE * 4);
    float* slik    = (float*)alloc((size_t)NSEDGE * 4);
    float* swgt    = (float*)alloc((size_t)NSEDGE * 4);
    int*   bidx    = (int*)  alloc((size_t)NBEDGE * 4);
    float* blik    = (float*)alloc((size_t)NBEDGE * 4);
    float* bw      = (float*)alloc((size_t)NBEDGE * 4);
    float* blog    = (float*)alloc((size_t)NBEDGE * 4);
    float* snin    = (float*)alloc((size_t)NCLUST * LATENT * 4);
    float* snA     = (float*)alloc((size_t)NCLUST * LATENT * 4);
    float* snB     = (float*)alloc((size_t)NCLUST * LATENT * 4);
    float* node2sn = (float*)alloc((size_t)NCLUST * LATENT * 4);
    float* se_agg  = (float*)alloc((size_t)NCLUST * LATENT * 4);
    float* seA     = (float*)alloc((size_t)NSEDGE * LATENT * 4);
    float* seB     = (float*)alloc((size_t)NSEDGE * LATENT * 4);
    float* nodesA  = (float*)alloc((size_t)NNODES * LATENT * 4);
    float* nodesB  = (float*)alloc((size_t)NNODES * LATENT * 4);
    float* edge_agg= (float*)alloc((size_t)NNODES * LATENT * 4);
    float* sn2node = (float*)alloc((size_t)NNODES * LATENT * 4);
    float* edgesA  = (float*)alloc((size_t)NEDGE * LATENT * 4);
    float* edgesB  = (float*)alloc((size_t)NEDGE * LATENT * 4);
    _Float16* H1   = (_Float16*)alloc((size_t)NEDGE * HIDDEN * 2);
    _Float16* H2   = (_Float16*)alloc((size_t)NEDGE * HIDDEN * 2);

    // ---- prepacked f16 transposed weights: Wt[Npad][K] ----
    static const int KL[6][3] = {
        {LATENT, HIDDEN, HIDDEN}, {2 * LATENT, HIDDEN, HIDDEN},
        {3 * LATENT, HIDDEN, HIDDEN}, {3 * LATENT, HIDDEN, HIDDEN},
        {3 * LATENT, HIDDEN, HIDDEN}, {3 * LATENT, HIDDEN, HIDDEN}};
    static const int NL[6][3] = {
        {HIDDEN, HIDDEN, LATENT - EMB}, {HIDDEN, HIDDEN, LATENT},
        {HIDDEN, HIDDEN, LATENT}, {HIDDEN, HIDDEN, LATENT},
        {HIDDEN, HIDDEN, LATENT}, {HIDDEN, HIDDEN, LATENT}};
    _Float16* WT[6][3];
    for (int mm = 0; mm < 6; ++mm)
        for (int l = 0; l < 3; ++l) {
            int K = KL[mm][l], N = NL[mm][l];
            int Npad = ((N + 15) / 16) * 16;
            WT[mm][l] = (_Float16*)alloc((size_t)Npad * K * 2);
            const float* W = (const float*)d_in[6 + mm * 12 + l * 4 + 0];
            k_prep_wt<<<gb((long long)Npad * K), 256, 0, stream>>>(W, K, N, Npad, WT[mm][l]);
        }

    // ---- zero accumulators ----
    hipMemsetAsync(means, 0, (size_t)NCLUST * EMB * 4, stream);
    hipMemsetAsync(counts, 0, (size_t)NCLUST * 4, stream);
    hipMemsetAsync(red, 0, 64, stream);
    hipMemsetAsync(snin, 0, (size_t)NCLUST * LATENT * 4, stream);

    // ---- cluster means (scatter_mean + unit-normalize) ----
    k_cluster_accum<<<gb(NNODES), 256, 0, stream>>>(emb, clusters, means, counts, NNODES);
    k_cluster_final<<<gb(NCLUST), 256, 0, stream>>>(means, counts);

    // ---- super graph: top-8 kNN on means, symmetric, sigmoid weights ----
    k_super_knn<<<4, 256, 0, stream>>>(means, sg0, sg1, slik);
    k_reduce2<<<64, 256, 0, stream>>>(slik, NSEDGE, red + 0);
    k_logits_w<<<64, 256, 0, stream>>>(slik, NSEDGE, red + 0, 1, nullptr, swgt, red + 2);
    k_w_norm<<<gb(NSEDGE), 256, 0, stream>>>(swgt, NSEDGE, red + 2);

    // ---- bipartite graph: top-4 kNN emb->means, exp weights ----
    k_bip_knn<<<gb(NNODES), 256, 0, stream>>>(emb, means, bidx, blik);
    k_reduce2<<<256, 256, 0, stream>>>(blik, NBEDGE, red + 4);
    k_logits_w<<<256, 256, 0, stream>>>(blik, NBEDGE, red + 4, 0, blog, bw, red + 6);
    k_w_norm<<<gb(NBEDGE), 256, 0, stream>>>(bw, NBEDGE, red + 6);

    // ---- supernode init: [means | MLP_sn_enc(sn_in)] ----
    k_sn_in<<<gb(NBEDGE), 256, 0, stream>>>(nodes_in, bidx, bw, snin, NBEDGE);
    k_copy_means<<<gb(NCLUST * EMB), 256, 0, stream>>>(means, snA);

    const float* Wp[12];
    getW(0, Wp);   // p_sn_enc: [128,256,256,104]
    {
        Cat3 c{snin, nullptr, nullptr, nullptr, nullptr, nullptr};
        run_mlp3(stream, NCLUST, c, LATENT, WT[0], Wp, H1, H2,
                 LATENT - EMB, snA + EMB, LATENT, nullptr, 0);
    }
    getW(1, Wp);   // p_se_enc: [256,256,256,128]
    {
        Cat3 c{snA, sg0, snA, sg1, nullptr, nullptr};
        run_mlp3(stream, NSEDGE, c, 2 * LATENT, WT[1], Wp, H1, H2,
                 LATENT, seA, LATENT, nullptr, 0);
    }

    // mutable copies of nodes/edges
    hipMemcpyAsync(nodesA, nodes_in, (size_t)NNODES * LATENT * 4, hipMemcpyDeviceToDevice, stream);
    hipMemcpyAsync(edgesA, edges_in, (size_t)NEDGE * LATENT * 4, hipMemcpyDeviceToDevice, stream);

    const float* Wn[12]; const float* We[12]; const float* Ws[12]; const float* Wse[12];
    getW(2, Wn); getW(3, We); getW(4, Ws); getW(5, Wse);

    float *nc = nodesA, *nn = nodesB, *sc = snA, *snw = snB;
    float *ec = edgesA, *en = edgesB, *sec = seA, *sen = seB;

    for (int it = 0; it < 2; ++it) {
        // nodes update
        hipMemsetAsync(edge_agg, 0, (size_t)NNODES * LATENT * 4, stream);
        k_scatter_add<<<gb((long long)NEDGE * LATENT), 256, 0, stream>>>(
            ec, nullptr, 0, nullptr, gr1, 0, edge_agg, (long long)NEDGE * LATENT);
        hipMemsetAsync(sn2node, 0, (size_t)NNODES * LATENT * 4, stream);
        k_scatter_add<<<gb((long long)NBEDGE * LATENT), 256, 0, stream>>>(
            sc, bidx, 0, bw, nullptr, 2, sn2node, (long long)NBEDGE * LATENT);
        {
            Cat3 c{nc, nullptr, edge_agg, nullptr, sn2node, nullptr};
            run_mlp3(stream, NNODES, c, 3 * LATENT, WT[2], Wn, H1, H2, LATENT, nn, LATENT, nc, LATENT);
        }
        // supernodes update (uses new nodes, old superedges)
        hipMemsetAsync(se_agg, 0, (size_t)NCLUST * LATENT * 4, stream);
        k_scatter_add<<<gb((long long)NSEDGE * LATENT), 256, 0, stream>>>(
            sec, nullptr, 0, swgt, sg1, 0, se_agg, (long long)NSEDGE * LATENT);
        hipMemsetAsync(node2sn, 0, (size_t)NCLUST * LATENT * 4, stream);
        k_scatter_add<<<gb((long long)NBEDGE * LATENT), 256, 0, stream>>>(
            nn, nullptr, 2, bw, bidx, 0, node2sn, (long long)NBEDGE * LATENT);
        {
            Cat3 c{sc, nullptr, node2sn, nullptr, se_agg, nullptr};
            run_mlp3(stream, NCLUST, c, 3 * LATENT, WT[4], Ws, H1, H2, LATENT, snw, LATENT, sc, LATENT);
        }
        // edges update (new nodes, old edges)
        {
            Cat3 c{nn, gr0, nn, gr1, ec, nullptr};
            run_mlp3(stream, NEDGE, c, 3 * LATENT, WT[3], We, H1, H2, LATENT, en, LATENT, ec, LATENT);
        }
        // superedges update (new supernodes, old superedges)
        {
            Cat3 c{snw, sg0, snw, sg1, sec, nullptr};
            run_mlp3(stream, NSEDGE, c, 3 * LATENT, WT[5], Wse, H1, H2, LATENT, sen, LATENT, sec, LATENT);
        }
        float* t;
        t = nc; nc = nn; nn = t;
        t = sc; sc = snw; snw = t;
        t = ec; ec = en; en = t;
        t = sec; sec = sen; sen = t;
    }

    // ---- outputs: nodes | supernodes | bip_graph(2x400000) | att_logits ----
    float* out = (float*)d_out;
    size_t o0 = 0;
    hipMemcpyAsync(out + o0, nc, (size_t)NNODES * LATENT * 4, hipMemcpyDeviceToDevice, stream);
    o0 += (size_t)NNODES * LATENT;
    hipMemcpyAsync(out + o0, sc, (size_t)NCLUST * LATENT * 4, hipMemcpyDeviceToDevice, stream);
    o0 += (size_t)NCLUST * LATENT;
    k_bip_out<<<gb(NBEDGE), 256, 0, stream>>>(bidx, out + o0, NBEDGE);
    o0 += (size_t)2 * NBEDGE;
    // att_logits = standardized logits (mean is analytically zero)
    hipMemcpyAsync(out + o0, blog, (size_t)NBEDGE * 4, hipMemcpyDeviceToDevice, stream);
}
